// FeatureMixer_71691594105027
// MI455X (gfx1250) — compile-verified, matrix operand order checked by reading, not run on hardware
//
#include <hip/hip_runtime.h>
#include <hip/hip_bf16.h>
#include <math.h>

// ---------------------------------------------------------------------------
// FeatureMixer for MI455X (gfx1250): bf16 WMMA (f32 accum) everywhere.
// Dims: B=4 Q=900 H=4 P=32 C=64 D1=64 EP=64  QUERY_DIM=256 TOTAL=24576
// B operands live in LDS fragment-packed (lane*16+elem) so each v_wmma B
// feed is two ds_load_b128; all staging moves data in 16B runs
// (8 consecutive K for fixed N map to one contiguous packed run).
// ---------------------------------------------------------------------------

typedef __attribute__((ext_vector_type(16))) __bf16 v16bf;
typedef __attribute__((ext_vector_type(8)))  __bf16 v8bf;
typedef __attribute__((ext_vector_type(8)))  float  v8f;

#define NQ      3600        // B*Q
#define QD      256         // query/feats dim
#define TOTAL   24576
#define P1SZ    16384       // H*C*D1 (== H*EP*D1 = mid2 size, region reused)

__device__ __forceinline__ v8f wmma_bf16(v16bf a, v16bf b, v8f c) {
  return __builtin_amdgcn_wmma_f32_16x16x32_bf16(false, a, false, b,
                                                 (short)0, c, false, false);
}

// A fragment: 16x32 (MxK) bf16, row-major tile, leading dim ld.
// Per-lane elements are two contiguous 16B runs -> 2x ds_load_b128.
__device__ __forceinline__ v16bf frag_a(const __bf16* tile, int ld) {
  const int lane = threadIdx.x & 31;
  const int m    = lane & 15;
  const int half = lane >> 4;
  v16bf f;
#pragma unroll
  for (int i = 0; i < 8; ++i) {
    const int k0 = ((i & 4) << 2) + (half << 3) + ((i & 3) << 1);
    f[2 * i]     = tile[m * ld + k0];
    f[2 * i + 1] = tile[m * ld + k0 + 1];
  }
  return f;
}

// Fragment-packed B tile: 512 bf16 laid out as [lane][elem] (32B per lane).
__device__ __forceinline__ v16bf frag_b_packed(const __bf16* buf) {
  const int lane = threadIdx.x & 31;
  return *(const v16bf*)(buf + lane * 16);
}

// Packed position of source element (k, n) of a 32x16 (KxN) B tile
// (used only by kernel 1's per-element epilogue remap).
__device__ __forceinline__ int bpack_off(int k, int n) {
  const int lane = n + ((k >> 3) & 1) * 16;
  const int i    = ((k & 16) >> 2) + ((k & 7) >> 1);
  return lane * 16 + 2 * i + (k & 1);
}

// Packed offset of the 8-element run covering k = g*8 .. g*8+7 at column n.
// (g&1) selects the lane half; (g&2) selects the second 16B run.
__device__ __forceinline__ int brun_off(int g, int n) {
  return (n >> 4) * 512 + ((n & 15) + (g & 1) * 16) * 16 + (g & 2) * 4;
}

__device__ __forceinline__ float gelu_exact(float x) {
  return 0.5f * x * (1.0f + erff(x * 0.70710678118654752440f));
}

// Async global->LDS 16B copy (CDNA5 async path, tracked by ASYNCcnt).
__device__ __forceinline__ void async_b128(unsigned lds_off,
                                           unsigned long long gaddr) {
  asm volatile("global_load_async_to_lds_b128 %0, %1, off"
               :: "v"(lds_off), "v"(gaddr) : "memory");
}
__device__ __forceinline__ void wait_async0() {
  asm volatile("s_wait_asynccnt 0x0" ::: "memory");
}

// ---------------------------------------------------------------------------
// Kernel 1: params = query_feats @ W_gen + b_gen -> bf16 workspace.
// grid = (24576/128, 3600/16), block = 256 (8 waves), 8 K-steps per wave.
// p1 region written FRAGMENT-PACKED per (h, ntile, ktile); p2 row-major.
// ---------------------------------------------------------------------------
__global__ void k_gen_gemm(const float* __restrict__ qf,
                           const float* __restrict__ Wg,
                           const float* __restrict__ bg,
                           __bf16* __restrict__ params) {
  extern __shared__ char smem[];
  __bf16* Al = (__bf16*)smem;            // 16 x 256 bf16  (8 KB)
  __bf16* Bl = (__bf16*)(smem + 8192);   // 8 packed tiles x 512 (8 KB)

  const int t    = threadIdx.x;
  const int q0   = blockIdx.y * 16;
  const int n0   = blockIdx.x * 128;
  const int wave = t >> 5;
  const int lane = t & 31;

  // Stage A rows in 16B runs: thread t owns row t>>4, 16-col segment t&15.
  {
    const int m = t >> 4, seg = (t & 15) * 16;
#pragma unroll
    for (int u = 0; u < 2; ++u) {
      v8bf pk;
#pragma unroll
      for (int j = 0; j < 8; ++j)
        pk[j] = (__bf16)qf[(size_t)(q0 + m) * QD + seg + u * 8 + j];
      *(v8bf*)(Al + m * QD + seg + u * 8) = pk;
    }
  }

  v8f acc = {};
  for (int kk = 0; kk < 8; ++kk) {
    // Prefetch next W_gen chunk (32x128 f32 = 128 cachelines).
    if (kk < 7 && t < 128) {
      const int kl = t >> 2, seg = t & 3;
      __builtin_prefetch(&Wg[(size_t)((kk + 1) * 32 + kl) * TOTAL + n0 + seg * 32], 0, 1);
    }
    __syncthreads();
    // Stage W_gen chunk (32 x 128) as packed 16B runs: thread t owns
    // K-group g = t>>6 (k = 8g..8g+7) and columns nl, nl+64.
    {
      const int g = t >> 6, nl = t & 63;
#pragma unroll
      for (int m2 = 0; m2 < 2; ++m2) {
        const int n = nl + m2 * 64;
        v8bf pk;
#pragma unroll
        for (int j = 0; j < 8; ++j)
          pk[j] = (__bf16)Wg[(size_t)(kk * 32 + g * 8 + j) * TOTAL + n0 + n];
        *(v8bf*)(Bl + brun_off(g, n)) = pk;
      }
    }
    __syncthreads();
    v16bf a = frag_a(Al + kk * 32, QD);
    v16bf b = frag_b_packed(Bl + wave * 512);
    acc = wmma_bf16(a, b, acc);
  }

  const int fn = lane & 15, half = lane >> 4;
  const int n = n0 + wave * 16 + fn;           // original column
  const float bias = bg[n];
  int off;                                     // remapped column in ws
  if (n < P1SZ) {                              // p1: fragment-packed
    const int h = n >> 12, c = (n >> 6) & 63, d = n & 63;
    off = (h * 8 + (d >> 4) * 2 + (c >> 5)) * 512 + bpack_off(c & 31, d & 15);
  } else {                                     // p2: row-major (A operand)
    off = n;
  }
#pragma unroll
  for (int r = 0; r < 8; ++r) {
    const int m = r + half * 8;
    params[(size_t)(q0 + m) * TOTAL + off] = (__bf16)(acc[r] + bias);
  }
}

// ---------------------------------------------------------------------------
// Kernel 2: per-query mixer. One block per (b,q), 256 threads (8 waves).
// LDS (~100 KB of the 320 KB WGP pool), aliased:
//   [0     ,16384) S (sampled bf16)      -> later M1 (normed, frag-packed)
//   [16384 ,49152) P1 (packed bf16)      \ later mid2 (f32 4x64x64 spans
//   [49152 ,81920) mid1 (f32 4x32x64)    /  both dead regions)
//   [81920 ,98304) P2 (bf16 4x64x32)
//   [98304 ,.....) reduction scratch + per-head stats
// p1/p2 staged from ws via global_load_async_to_lds_b128 (ASYNCcnt).
// ---------------------------------------------------------------------------
__global__ void k_mixer(const float* __restrict__ sf,
                        __bf16* __restrict__ params) {
  extern __shared__ char smem[];
  __bf16* S    = (__bf16*)smem;
  __bf16* M1   = (__bf16*)smem;             // alias of S (S dead by then)
  __bf16* P1   = (__bf16*)(smem + 16384);
  float*  mid2 = (float*)(smem + 16384);    // alias of P1+mid1 (both dead)
  float*  mid1 = (float*)(smem + 49152);
  __bf16* P2   = (__bf16*)(smem + 81920);
  float*  redS = (float*)(smem + 98304);
  float*  redQ = (float*)(smem + 99328);
  float*  stat = (float*)(smem + 100352);

  const int t  = threadIdx.x;
  const int bq = blockIdx.x;
  const size_t pbase = (size_t)bq * TOTAL;

  // Async-stage p1 (32 KB) and p2 (16 KB) bf16 tiles straight into LDS.
  {
    const unsigned p1_lds = (unsigned)(size_t)(void*)P1;
    const unsigned p2_lds = (unsigned)(size_t)(void*)P2;
    const unsigned long long gp = (unsigned long long)(const void*)(params + pbase);
#pragma unroll
    for (int i = 0; i < 8; ++i)
      async_b128(p1_lds + (t + i * 256) * 16,
                 gp + (unsigned long long)(t + i * 256) * 16);
#pragma unroll
    for (int i = 0; i < 4; ++i)
      async_b128(p2_lds + (t + i * 256) * 16,
                 gp + 32768ull + (unsigned long long)(t + i * 256) * 16);
  }
  // Overlap: stage sampled features (f32->bf16) in per-thread 128B spans.
  {
    const float* src = sf + (size_t)bq * 8192 + t * 32;
#pragma unroll
    for (int i = 0; i < 4; ++i) {
      v8bf pk;
#pragma unroll
      for (int j = 0; j < 8; ++j) pk[j] = (__bf16)src[i * 8 + j];
      *(v8bf*)(S + t * 32 + i * 8) = pk;
    }
  }
  wait_async0();
  __syncthreads();

  const int wave = t >> 5, lane = t & 31;
  const int fn = lane & 15, half = lane >> 4;

  // ---- Phase 1: per-head (32x64) = S_h(32x64) @ P1_h(64x64), GELU ----
#pragma unroll
  for (int i = 0; i < 4; ++i) {
    const int tt = wave + (i << 3);
    const int h = tt >> 3, mt = (tt >> 2) & 1, nt = tt & 3;
    v8f acc = {};
#pragma unroll
    for (int kk = 0; kk < 2; ++kk) {
      v16bf a = frag_a(S + h * 2048 + mt * 1024 + kk * 32, 64);
      v16bf b = frag_b_packed(P1 + (h * 8 + nt * 2 + kk) * 512);
      acc = wmma_bf16(a, b, acc);
    }
#pragma unroll
    for (int r = 0; r < 8; ++r) {
      const int m = mt * 16 + r + half * 8;
      mid1[h * 2048 + m * 64 + nt * 16 + fn] = gelu_exact(acc[r]);
    }
  }
  __syncthreads();

  // ---- LayerNorm over (32x64) per head; write M1 fragment-packed ----
  {
    const int h = t >> 6, j = t & 63;
    float s = 0.f, q = 0.f;
#pragma unroll 8
    for (int i = 0; i < 32; ++i) {
      const float v = mid1[h * 2048 + j + i * 64];
      s += v; q += v * v;
    }
    redS[t] = s; redQ[t] = q;
    __syncthreads();
    if (j == 0) {
      float ss = 0.f, qq = 0.f;
      for (int k = 0; k < 64; ++k) { ss += redS[h * 64 + k]; qq += redQ[h * 64 + k]; }
      const float mean = ss * (1.f / 2048.f);
      const float var  = qq * (1.f / 2048.f) - mean * mean;
      stat[2 * h] = mean; stat[2 * h + 1] = rsqrtf(var + 1e-5f);
    }
    __syncthreads();
    // Normalize + pack in 16B runs: run r covers p = 8g..8g+7 at col n16.
#pragma unroll
    for (int i = 0; i < 4; ++i) {
      const int r = t + i * 256;          // 0..1023
      const int g = r >> 8;               // p-group
      const int tile = (r >> 4) & 15;     // h*4 + nt
      const int n16 = r & 15;
      const int hh = tile >> 2, nt = tile & 3;
      const float mean = stat[2 * hh], rstd = stat[2 * hh + 1];
      v8bf pk;
#pragma unroll
      for (int j2 = 0; j2 < 8; ++j2)
        pk[j2] = (__bf16)((mid1[hh * 2048 + (g * 8 + j2) * 64 + nt * 16 + n16]
                           - mean) * rstd);
      *(v8bf*)(M1 + tile * 512 + ((n16 + (g & 1) * 16) << 4) + (g & 2) * 4) = pk;
    }
  }
  __syncthreads();

  // ---- Phase 2: per-head (64x64) = P2_h(64x32) @ M1_h(32x64), GELU ----
#pragma unroll
  for (int i = 0; i < 8; ++i) {
    const int tt = wave + (i << 3);
    const int h = tt >> 4, mt = (tt >> 2) & 3, nt = tt & 3;
    v16bf a = frag_a(P2 + h * 2048 + mt * 512, 32);
    v16bf b = frag_b_packed(M1 + (h * 4 + nt) * 512);
    v8f acc = {};
    acc = wmma_bf16(a, b, acc);
#pragma unroll
    for (int r = 0; r < 8; ++r) {
      const int m = mt * 16 + r + half * 8;
      mid2[h * 4096 + m * 64 + nt * 16 + fn] = gelu_exact(acc[r]);
    }
  }
  __syncthreads();

  // ---- LayerNorm over (64x64) per head; bf16 out over dead p1 region ----
  // (row-major (h,e,d): the A operand of kernel 3)
  {
    const int h = t >> 6, j = t & 63;
    float s = 0.f, q = 0.f;
    for (int i = 0; i < 64; ++i) {
      const float v = mid2[h * 4096 + j + i * 64];
      s += v; q += v * v;
    }
    redS[t] = s; redQ[t] = q;
    __syncthreads();
    if (j == 0) {
      float ss = 0.f, qq = 0.f;
      for (int k = 0; k < 64; ++k) { ss += redS[h * 64 + k]; qq += redQ[h * 64 + k]; }
      const float mean = ss * (1.f / 4096.f);
      const float var  = qq * (1.f / 4096.f) - mean * mean;
      stat[2 * h] = mean; stat[2 * h + 1] = rsqrtf(var + 1e-5f);
    }
    __syncthreads();
    // Thread t owns 64 contiguous elems (single head) -> 8x b128 stores.
    {
      const int hh = t >> 6;
      const float mean = stat[2 * hh], rstd = stat[2 * hh + 1];
#pragma unroll
      for (int i = 0; i < 8; ++i) {
        v8bf pk;
#pragma unroll
        for (int j2 = 0; j2 < 8; ++j2)
          pk[j2] = (__bf16)((mid2[t * 64 + i * 8 + j2] - mean) * rstd);
        *(v8bf*)(params + pbase + t * 64 + i * 8) = pk;
      }
    }
  }
}

// ---------------------------------------------------------------------------
// Kernel 3: out = LN( mid2(3600x16384) @ W_out(16384x256) + b_out ) * w + b
// grid = 225, block = 256 (8 waves). Wave w owns cols [32w,32w+32): 2 tiles.
// K streamed in 512 chunks of 32; A chunk async-copied, W_out chunk
// converted+packed as 16B runs (L2-resident at 16.8 MB).
// ---------------------------------------------------------------------------
__global__ void k_out_proj(const __bf16* __restrict__ params,
                           const float* __restrict__ Wo,
                           const float* __restrict__ bo,
                           const float* __restrict__ lnw,
                           const float* __restrict__ lnb,
                           float* __restrict__ out) {
  extern __shared__ char smem[];
  __bf16* Al   = (__bf16*)smem;                    // 16 x 32 bf16 (1 KB)
  __bf16* Bl   = (__bf16*)(smem + 1024);           // 16 packed tiles (16 KB)
  float*  outb = (float*)(smem + 17408);           // 16 x 256 f32 (16 KB)
  float*  redS = (float*)(smem + 33792);
  float*  redQ = (float*)(smem + 34816);
  float*  stat = (float*)(smem + 35840);

  const int t  = threadIdx.x;
  const int q0 = blockIdx.x * 16;
  const int wave = t >> 5, lane = t & 31;
  const int fn = lane & 15, half = lane >> 4;
  const unsigned al_lds = (unsigned)(size_t)(void*)Al;

  v8f acc0 = {}, acc1 = {};
  for (int kc = 0; kc < 512; ++kc) {
    // Prefetch next W_out chunk (32x256 f32 = 32 KB = 256 cachelines).
    if (kc < 511)
      __builtin_prefetch(&Wo[(size_t)(kc + 1) * 8192 + t * 32], 0, 1);
    __syncthreads();
    // A chunk: 16x32 bf16 async-copied from the mixer-output ws region.
    if (t < 64) {
      const int m = t >> 2, kseg = (t & 3) * 8;
      async_b128(al_lds + (unsigned)t * 16,
                 (unsigned long long)(const void*)
                     (params + (size_t)(q0 + m) * TOTAL + kc * 32 + kseg));
    }
    // B chunk: 32x256 f32 -> bf16 in packed 16B runs: thread t owns
    // K-group g = t>>6 and columns nl + {0,64,128,192}.
    {
      const int g = t >> 6, nl = t & 63;
#pragma unroll
      for (int m2 = 0; m2 < 4; ++m2) {
        const int n = nl + m2 * 64;
        v8bf pk;
#pragma unroll
        for (int j = 0; j < 8; ++j)
          pk[j] = (__bf16)Wo[(size_t)(kc * 32 + g * 8 + j) * QD + n];
        *(v8bf*)(Bl + brun_off(g, n)) = pk;
      }
    }
    wait_async0();
    __syncthreads();
    v16bf a  = frag_a(Al, 32);
    v16bf b0 = frag_b_packed(Bl + (wave * 2) * 512);
    v16bf b1 = frag_b_packed(Bl + (wave * 2 + 1) * 512);
    acc0 = wmma_bf16(a, b0, acc0);
    acc1 = wmma_bf16(a, b1, acc1);
  }
  __syncthreads();

  // Epilogue: bias -> LDS, affine row LayerNorm, coalesced store.
#pragma unroll
  for (int r = 0; r < 8; ++r) {
    const int m = r + half * 8;
    const int c0 = wave * 32 + fn;
    const int c1 = wave * 32 + 16 + fn;
    outb[m * 256 + c0] = acc0[r] + bo[c0];
    outb[m * 256 + c1] = acc1[r] + bo[c1];
  }
  __syncthreads();
  {
    const int row = t >> 4, s4 = t & 15;
    float s = 0.f, q = 0.f;
#pragma unroll
    for (int i = 0; i < 16; ++i) {
      const float v = outb[row * 256 + s4 + i * 16];
      s += v; q += v * v;
    }
    redS[t] = s; redQ[t] = q;
    __syncthreads();
    if (s4 == 0) {
      float ss = 0.f, qq = 0.f;
      for (int k = 0; k < 16; ++k) { ss += redS[row * 16 + k]; qq += redQ[row * 16 + k]; }
      const float mean = ss * (1.f / 256.f);
      const float var  = qq * (1.f / 256.f) - mean * mean;
      stat[2 * row] = mean; stat[2 * row + 1] = rsqrtf(var + 1e-5f);
    }
    __syncthreads();
#pragma unroll
    for (int i = 0; i < 16; ++i) {
      const int idx = t + i * 256;
      const int row2 = idx >> 8, col = idx & 255;
      const float v = (outb[idx] - stat[2 * row2]) * stat[2 * row2 + 1];
      out[(size_t)(q0 + row2) * QD + col] = v * lnw[col] + lnb[col];
    }
  }
}

// ---------------------------------------------------------------------------
// Launch
// ---------------------------------------------------------------------------
extern "C" void kernel_launch(void* const* d_in, const int* in_sizes, int n_in,
                              void* d_out, int out_size, void* d_ws, size_t ws_size,
                              hipStream_t stream) {
  const float* qf  = (const float*)d_in[0];
  const float* sf  = (const float*)d_in[1];
  const float* Wg  = (const float*)d_in[2];
  const float* bg  = (const float*)d_in[3];
  const float* Wo  = (const float*)d_in[4];
  const float* bo  = (const float*)d_in[5];
  const float* lnw = (const float*)d_in[6];
  const float* lnb = (const float*)d_in[7];
  float* outp = (float*)d_out;

  __bf16* params = (__bf16*)d_ws;   // NQ * TOTAL bf16 = ~170 MB

  dim3 blk(256);
  k_gen_gemm<<<dim3(TOTAL / 128, NQ / 16), blk, 16384, stream>>>(qf, Wg, bg, params);
  k_mixer<<<dim3(NQ), blk, 100480, stream>>>(sf, params);
  k_out_proj<<<dim3(NQ / 16), blk, 35968, stream>>>(params, Wo, bo, lnw, lnb, outp);
}
